// QuantizedLinear_20375324852409
// MI455X (gfx1250) — 8-bit matrix operand order repaired and checked by reading, compiled, not run on hardware
//
#include <hip/hip_runtime.h>
#include <hip/hip_bf16.h>

typedef int v8i __attribute__((ext_vector_type(8)));

#define M_DIM 32
#define K_DIM 8192
#define N_DIM 14336
#define NGROUPS 64
#define GSIZE 128

__global__ __launch_bounds__(256) void quant_act_kernel(
    const float* __restrict__ x,
    signed char* __restrict__ qx,
    float* __restrict__ scale_out,
    int* __restrict__ zp_out)
{
    const int row = blockIdx.x;
    const int tid = threadIdx.x;
    const float* xr = x + (size_t)row * K_DIM;

    float vmin = 3.402823466e38f, vmax = -3.402823466e38f;
    for (int i = tid; i < K_DIM; i += 256) {
        float v = xr[i];
        vmin = fminf(vmin, v);
        vmax = fmaxf(vmax, v);
    }
    __shared__ float smin[256];
    __shared__ float smax[256];
    smin[tid] = vmin; smax[tid] = vmax;
    __syncthreads();
    for (int s = 128; s > 0; s >>= 1) {
        if (tid < s) {
            smin[tid] = fminf(smin[tid], smin[tid + s]);
            smax[tid] = fmaxf(smax[tid], smax[tid + s]);
        }
        __syncthreads();
    }
    __shared__ float s_scale, s_zp;
    if (tid == 0) {
        float xmin = smin[0], xmax = smax[0];
        float sc = fmaxf((xmax - xmin) * (1.0f / 255.0f), 1e-8f);
        float zp = __builtin_rintf(-128.0f - xmin / sc);
        zp = fminf(fmaxf(zp, -128.0f), 127.0f);
        s_scale = sc; s_zp = zp;
        scale_out[row] = sc;
        zp_out[row] = (int)zp;
    }
    __syncthreads();
    float sc = s_scale, zp = s_zp;
    for (int i = tid; i < K_DIM; i += 256) {
        float q = __builtin_rintf(xr[i] / sc) + zp;
        q = fminf(fmaxf(q, -128.0f), 127.0f);
        qx[(size_t)row * K_DIM + i] = (signed char)(int)q;
    }
}

__device__ __forceinline__ int pack4_sub8(int4 q) {
    unsigned b0 = (unsigned)(q.x - 8) & 0xFFu;
    unsigned b1 = (unsigned)(q.y - 8) & 0xFFu;
    unsigned b2 = (unsigned)(q.z - 8) & 0xFFu;
    unsigned b3 = (unsigned)(q.w - 8) & 0xFFu;
    return (int)(b0 | (b1 << 8) | (b2 << 16) | (b3 << 24));
}

__global__ __launch_bounds__(256) void qlin_wmma_kernel(
    const int* __restrict__ qw,
    const float* __restrict__ wsc,
    const signed char* __restrict__ qx,
    const float* __restrict__ scale,
    const int* __restrict__ zp,
    float* __restrict__ pout,
    int groups_per_chunk,
    const float* __restrict__ bias)
{
    const int wave  = threadIdx.x >> 5;
    const int lane  = threadIdx.x & 31;
    const int col   = lane & 15;
    const int khalf = lane >> 4;
    const int n0 = (blockIdx.x * 8 + wave) * 16;
    const int n  = n0 + col;
    const int chunk = blockIdx.y;
    const int g0 = chunk * groups_per_chunk;
    const int g1 = g0 + groups_per_chunk;

    const int*   qwrow = qw  + (size_t)n * K_DIM;
    const float* wsrow = wsc + (size_t)n * NGROUPS;
    const signed char* qxa = qx + (size_t)(col) * K_DIM;
    const signed char* qxb = qx + (size_t)(16 + col) * K_DIM;

    int zpa[8], zpb[8];
#pragma unroll
    for (int r = 0; r < 8; ++r) {
        zpa[r] = zp[r + 8 * khalf];
        zpb[r] = zp[16 + r + 8 * khalf];
    }

    v8i onesA;
#pragma unroll
    for (int i = 0; i < 8; ++i) onesA[i] = 0x01010101;

    float acc0[8], acc1[8];
#pragma unroll
    for (int r = 0; r < 8; ++r) { acc0[r] = 0.0f; acc1[r] = 0.0f; }

    for (int g = g0; g < g1; ++g) {
        const int kb = g * GSIZE;
        __builtin_prefetch(qwrow + kb + GSIZE + khalf * 16, 0, 1);

        v8i c0 = {}, c1 = {}, cs = {};
#pragma unroll
        for (int s = 0; s < 2; ++s) {
            const int kbs = kb + 64 * s;

            v8i bf;
            {
                const int* p = qwrow + kbs + khalf * 16;
                int4 q;
                q = *(const int4*)(p +  0); bf[0] = pack4_sub8(q);
                q = *(const int4*)(p +  4); bf[1] = pack4_sub8(q);
                q = *(const int4*)(p +  8); bf[2] = pack4_sub8(q);
                q = *(const int4*)(p + 12); bf[3] = pack4_sub8(q);
                p = qwrow + kbs + 32 + khalf * 16;
                q = *(const int4*)(p +  0); bf[4] = pack4_sub8(q);
                q = *(const int4*)(p +  4); bf[5] = pack4_sub8(q);
                q = *(const int4*)(p +  8); bf[6] = pack4_sub8(q);
                q = *(const int4*)(p + 12); bf[7] = pack4_sub8(q);
            }

            v8i a0, a1;
            {
                const int* p = (const int*)(qxa + kbs);
                int2 t;
                t = *(const int2*)(p +  0 + 4 * khalf); a0[0] = t.x; a0[1] = t.y;
                t = *(const int2*)(p +  2 + 4 * khalf); a0[2] = t.x; a0[3] = t.y;
                t = *(const int2*)(p +  8 + 4 * khalf); a0[4] = t.x; a0[5] = t.y;
                t = *(const int2*)(p + 10 + 4 * khalf); a0[6] = t.x; a0[7] = t.y;
                p = (const int*)(qxb + kbs);
                t = *(const int2*)(p +  0 + 4 * khalf); a1[0] = t.x; a1[1] = t.y;
                t = *(const int2*)(p +  2 + 4 * khalf); a1[2] = t.x; a1[3] = t.y;
                t = *(const int2*)(p +  8 + 4 * khalf); a1[4] = t.x; a1[5] = t.y;
                t = *(const int2*)(p + 10 + 4 * khalf); a1[6] = t.x; a1[7] = t.y;
            }

            c0 = __builtin_amdgcn_wmma_i32_16x16x64_iu8(true, a0,    true, bf, c0, false, false);
            c1 = __builtin_amdgcn_wmma_i32_16x16x64_iu8(true, a1,    true, bf, c1, false, false);
            cs = __builtin_amdgcn_wmma_i32_16x16x64_iu8(true, onesA, true, bf, cs, false, false);
        }

        const float w = wsrow[g];
        const int csum = cs[0];
#pragma unroll
        for (int r = 0; r < 8; ++r) {
            acc0[r] += w * (float)(c0[r] - zpa[r] * csum);
            acc1[r] += w * (float)(c1[r] - zpb[r] * csum);
        }
    }

    float* outc = pout + (size_t)chunk * M_DIM * N_DIM;
    const float bv = bias ? bias[n] : 0.0f;
#pragma unroll
    for (int r = 0; r < 8; ++r) {
        const int m0 = r + 8 * khalf;
        const int m1 = 16 + r + 8 * khalf;
        outc[(size_t)m0 * N_DIM + n] = scale[m0] * acc0[r] + bv;
        outc[(size_t)m1 * N_DIM + n] = scale[m1] * acc1[r] + bv;
    }
}

__global__ __launch_bounds__(256) void reduce_kernel(
    const float* __restrict__ partial,
    const float* __restrict__ bias,
    float* __restrict__ out,
    int nchunks)
{
    const int idx = blockIdx.x * 256 + threadIdx.x;
    if (idx >= M_DIM * N_DIM) return;
    const int n = idx % N_DIM;
    float s = bias[n];
    for (int c = 0; c < nchunks; ++c)
        s += partial[(size_t)c * M_DIM * N_DIM + idx];
    out[idx] = s;
}

extern "C" void kernel_launch(void* const* d_in, const int* in_sizes, int n_in,
                              void* d_out, int out_size, void* d_ws, size_t ws_size,
                              hipStream_t stream) {
    const float* x    = (const float*)d_in[0];
    const int*   qw   = (const int*)d_in[1];
    const float* wsc  = (const float*)d_in[2];
    const float* bias = (const float*)d_in[3];

    signed char* qx = (signed char*)d_ws;
    float* scale = (float*)((char*)d_ws + (size_t)M_DIM * K_DIM);
    int*   zp    = (int*)((char*)d_ws + (size_t)M_DIM * K_DIM + 128);
    const size_t base = (size_t)M_DIM * K_DIM + 256;
    float* partial = (float*)((char*)d_ws + base);
    const size_t per_chunk = (size_t)M_DIM * N_DIM * sizeof(float);

    quant_act_kernel<<<M_DIM, 256, 0, stream>>>(x, qx, scale, zp);

    int nchunks = 4;
    if (ws_size < base + (size_t)nchunks * per_chunk) nchunks = 1;
    if (ws_size < base + per_chunk) nchunks = 0;

    const int nblk = N_DIM / (8 * 16);
    if (nchunks == 0) {
        qlin_wmma_kernel<<<dim3(nblk, 1), 256, 0, stream>>>(
            qw, wsc, qx, scale, zp, (float*)d_out, NGROUPS, bias);
    } else {
        qlin_wmma_kernel<<<dim3(nblk, nchunks), 256, 0, stream>>>(
            qw, wsc, qx, scale, zp, partial, NGROUPS / nchunks, nullptr);
        reduce_kernel<<<(M_DIM * N_DIM + 255) / 256, 256, 0, stream>>>(
            partial, bias, (float*)d_out, nchunks);
    }
}
